// KLayer_62551903698991
// MI455X (gfx1250) — matrix operand order repaired and checked by reading, not compiled, not run on hardware
//
#include <hip/hip_runtime.h>

// ---------------------------------------------------------------------------
// KAN layer for MI455X (gfx1250):
//   out = silu(x) @ base_weight.T + bspline_basis(x) @ (spline_weight*scaler).T
// Fused as one bf16 WMMA GEMM with K = 1024 + 1024*8 = 9216.
// GEMM uses async global->LDS staging (ASYNCcnt) with double buffering.
// ---------------------------------------------------------------------------

#define IN_F   1024
#define OUT_F  1024
#define BATCH  4096
#define NCOEF  8                       // GRID_SIZE + SPLINE_ORDER
#define KTOT   (IN_F + IN_F * NCOEF)   // 9216

typedef __attribute__((ext_vector_type(16))) __bf16 v16bf;
typedef __attribute__((ext_vector_type(8)))  __bf16 v8bf;
typedef __attribute__((ext_vector_type(8)))  float  v8f;

// ---------------------------------------------------------------------------
// Kernel 1: build bf16 weight matrix Wb (OUT_F x KTOT), K-major.
// ---------------------------------------------------------------------------
__global__ __launch_bounds__(256) void kan_prep_w(
    const float* __restrict__ base_w,
    const float* __restrict__ spline_w,
    const float* __restrict__ spline_s,
    __bf16* __restrict__ Wb)
{
    int idx = blockIdx.x * blockDim.x + threadIdx.x;   // one (n, i) pair
    if (idx >= OUT_F * IN_F) return;
    int n = idx / IN_F;
    int i = idx - n * IN_F;

    size_t wrow = (size_t)n * KTOT;
    Wb[wrow + i] = (__bf16)base_w[idx];

    float sc = spline_s[idx];
    const float* sw = spline_w + (size_t)idx * NCOEF;
    __bf16* dst = Wb + wrow + IN_F + (size_t)i * NCOEF;
#pragma unroll
    for (int c = 0; c < NCOEF; ++c)
        dst[c] = (__bf16)(sw[c] * sc);
}

// ---------------------------------------------------------------------------
// Kernel 2: build bf16 activation matrix Ab (BATCH x KTOT), K-major.
//   Ab[m][i]              = silu(x[m][i])
//   Ab[m][1024 + i*8 + c] = cubic B-spline basis_c(x[m][i])   (Cox-de Boor)
// ---------------------------------------------------------------------------
__global__ __launch_bounds__(256) void kan_prep_a(
    const float* __restrict__ x,
    const float* __restrict__ grid,     // (IN_F, 12)
    __bf16* __restrict__ Ab)
{
    int idx = blockIdx.x * blockDim.x + threadIdx.x;   // one (m, i) pair
    if (idx >= BATCH * IN_F) return;
    int m = idx / IN_F;
    int i = idx - m * IN_F;

    float xv = x[idx];

    float t[12];
    const float* g = grid + (size_t)i * 12;
#pragma unroll
    for (int j = 0; j < 12; ++j) t[j] = g[j];

    float bas[11];
#pragma unroll
    for (int j = 0; j < 11; ++j)
        bas[j] = (xv >= t[j] && xv < t[j + 1]) ? 1.0f : 0.0f;

#pragma unroll
    for (int k = 1; k <= 3; ++k) {
#pragma unroll
        for (int j = 0; j < 11; ++j) {
            if (j + k < 11) {
                float t1 = (xv - t[j]) / (t[j + k] - t[j] + 1e-8f) * bas[j];
                float t2 = (t[j + k + 1] - xv) / (t[j + k + 1] - t[j + 1] + 1e-8f) * bas[j + 1];
                bas[j] = t1 + t2;
            }
        }
    }

    size_t arow = (size_t)m * KTOT;
    float sg = 1.0f / (1.0f + expf(-xv));
    Ab[arow + i] = (__bf16)(xv * sg);

    __bf16* dst = Ab + arow + IN_F + (size_t)i * NCOEF;
#pragma unroll
    for (int c = 0; c < NCOEF; ++c)
        dst[c] = (__bf16)bas[c];
}

// ---------------------------------------------------------------------------
// Async global -> LDS helpers (CDNA5 ASYNCcnt path)
// ---------------------------------------------------------------------------
__device__ __forceinline__ void async_copy_b128(unsigned lds_off, const void* gptr) {
    // per-lane: LDS[lds_off] = GLOBAL[gptr], 16 bytes, tracked by ASYNCcnt
    asm volatile("global_load_async_to_lds_b128 %0, %1, off"
                 :: "v"(lds_off), "v"((unsigned long long)(uintptr_t)gptr)
                 : "memory");
}

__device__ __forceinline__ void wait_async0() {
#if __has_builtin(__builtin_amdgcn_s_wait_asynccnt)
    __builtin_amdgcn_s_wait_asynccnt(0);
#else
    asm volatile("s_wait_asynccnt 0x0" ::: "memory");
#endif
}

// ---------------------------------------------------------------------------
// Kernel 3: C(M,N) = A(M,K) * W(N,K)^T  with v_wmma_f32_16x16x32_bf16.
// Block 128x128, BK=32, 256 threads = 8 waves (2 M x 4 N), 64x32 per wave.
// Double-buffered LDS, async staging overlapped with WMMA compute.
// ---------------------------------------------------------------------------
#define BM 128
#define BN 128
#define BK 32

__global__ __launch_bounds__(256) void kan_gemm(
    const __bf16* __restrict__ A,   // (BATCH, KTOT)
    const __bf16* __restrict__ W,   // (OUT_F, KTOT)
    float* __restrict__ C)          // (BATCH, OUT_F)
{
    __shared__ __bf16 lsa[2][BM][BK];   // 2 x 8 KB
    __shared__ __bf16 lsb[2][BN][BK];   // 2 x 8 KB

    const int tid  = threadIdx.x;
    const int lane = tid & 31;
    const int wave = tid >> 5;
    const int wm   = wave >> 2;     // 0..1  -> M offset wm*64
    const int wn   = wave & 3;      // 0..3  -> N offset wn*32
    const int m0   = blockIdx.y * BM;
    const int n0   = blockIdx.x * BN;

    const int lrow  = lane & 15;
    const int lhalf = lane >> 4;    // 0 or 1

    // staging geometry: 512 chunks of 8 bf16 (16B) per 128x32 tile, 2/thread
    const int srow0 = tid >> 2;                 // chunk tid
    const int sko0  = (tid & 3) << 3;
    const int srow1 = (tid + 256) >> 2;         // chunk tid+256
    const int sko1  = ((tid + 256) & 3) << 3;

    const __bf16* arow0 = A + (size_t)(m0 + srow0) * KTOT + sko0;
    const __bf16* arow1 = A + (size_t)(m0 + srow1) * KTOT + sko1;
    const __bf16* brow0 = W + (size_t)(n0 + srow0) * KTOT + sko0;
    const __bf16* brow1 = W + (size_t)(n0 + srow1) * KTOT + sko1;

    v8f acc[4][2];
#pragma unroll
    for (int mi = 0; mi < 4; ++mi)
#pragma unroll
        for (int ni = 0; ni < 2; ++ni)
            acc[mi][ni] = (v8f){0.f, 0.f, 0.f, 0.f, 0.f, 0.f, 0.f, 0.f};

    // prologue: stage tile kt=0 into buffer 0
    async_copy_b128((unsigned)(uintptr_t)&lsa[0][srow0][sko0], arow0);
    async_copy_b128((unsigned)(uintptr_t)&lsa[0][srow1][sko1], arow1);
    async_copy_b128((unsigned)(uintptr_t)&lsb[0][srow0][sko0], brow0);
    async_copy_b128((unsigned)(uintptr_t)&lsb[0][srow1][sko1], brow1);

    int buf = 0;
    for (int kt = 0; kt < KTOT; kt += BK) {
        wait_async0();        // this wave's tile-kt async copies done
        __syncthreads();      // => every wave's copies done; buf readable

        // stage next tile into the other buffer (async; overlaps compute).
        // Last iteration re-stages the final tile (never read) to stay
        // branch-free in the hot loop.
        int kn  = (kt + BK < KTOT) ? (kt + BK) : kt;
        int nb  = buf ^ 1;
        async_copy_b128((unsigned)(uintptr_t)&lsa[nb][srow0][sko0], arow0 + kn);
        async_copy_b128((unsigned)(uintptr_t)&lsa[nb][srow1][sko1], arow1 + kn);
        async_copy_b128((unsigned)(uintptr_t)&lsb[nb][srow0][sko0], brow0 + kn);
        async_copy_b128((unsigned)(uintptr_t)&lsb[nb][srow1][sko1], brow1 + kn);

        // A fragments: lane L holds M = L%16; K = {h*8..h*8+7} U {h*8+16..}
        v16bf af[4];
#pragma unroll
        for (int mi = 0; mi < 4; ++mi) {
            const __bf16* p = &lsa[buf][wm * 64 + mi * 16 + lrow][lhalf * 8];
            v8bf lo = *(const v8bf*)p;
            v8bf hi = *(const v8bf*)(p + 16);
#pragma unroll
            for (int e = 0; e < 8; ++e) { af[mi][e] = lo[e]; af[mi][8 + e] = hi[e]; }
        }

        // B fragments: lane L holds N = L%16; K = {h*8..h*8+7} U {h*8+16..}
        v16bf bfq[2];
#pragma unroll
        for (int ni = 0; ni < 2; ++ni) {
            const __bf16* p = &lsb[buf][wn * 32 + ni * 16 + lrow][lhalf * 8];
            v8bf lo = *(const v8bf*)p;
            v8bf hi = *(const v8bf*)(p + 16);
#pragma unroll
            for (int e = 0; e < 8; ++e) { bfq[ni][e] = lo[e]; bfq[ni][8 + e] = hi[e]; }
        }

#pragma unroll
        for (int mi = 0; mi < 4; ++mi)
#pragma unroll
            for (int ni = 0; ni < 2; ++ni)
                acc[mi][ni] = __builtin_amdgcn_wmma_f32_16x16x32_bf16(
                    /*neg_a=*/false, af[mi], /*neg_b=*/false, bfq[ni],
                    /*c_mod=*/(short)0, acc[mi][ni],
                    /*reuse_a=*/false, /*reuse_b=*/false);

        __syncthreads();      // all waves done reading buf before reuse
        buf ^= 1;
    }

    // C/D layout: VGPR v -> M = v + 8*(lane>=16); N = lane%16
#pragma unroll
    for (int mi = 0; mi < 4; ++mi) {
#pragma unroll
        for (int ni = 0; ni < 2; ++ni) {
            int rbase = m0 + wm * 64 + mi * 16 + lhalf * 8;
            int col   = n0 + wn * 32 + ni * 16 + lrow;
#pragma unroll
            for (int v = 0; v < 8; ++v)
                C[(size_t)(rbase + v) * OUT_F + col] = acc[mi][ni][v];
        }
    }
}

// ---------------------------------------------------------------------------
// Launcher
// ---------------------------------------------------------------------------
extern "C" void kernel_launch(void* const* d_in, const int* in_sizes, int n_in,
                              void* d_out, int out_size, void* d_ws, size_t ws_size,
                              hipStream_t stream) {
    const float* x        = (const float*)d_in[0];   // (4096, 1024)
    const float* base_w   = (const float*)d_in[1];   // (1024, 1024)
    const float* spline_w = (const float*)d_in[2];   // (1024, 1024, 8)
    const float* spline_s = (const float*)d_in[3];   // (1024, 1024)
    const float* grid     = (const float*)d_in[4];   // (1024, 12)
    float* out = (float*)d_out;                      // (4096, 1024)

    __bf16* Wb = (__bf16*)d_ws;                      // OUT_F * KTOT  (18.9 MB)
    __bf16* Ab = Wb + (size_t)OUT_F * KTOT;          // BATCH * KTOT  (75.5 MB)

    kan_prep_w<<<(OUT_F * IN_F + 255) / 256, 256, 0, stream>>>(base_w, spline_w, spline_s, Wb);
    kan_prep_a<<<(BATCH * IN_F + 255) / 256, 256, 0, stream>>>(x, grid, Ab);

    dim3 g(OUT_F / BN, BATCH / BM);   // (8, 32)
    kan_gemm<<<g, 256, 0, stream>>>(Ab, Wb, out);
}
